// Attention_36679020707895
// MI455X (gfx1250) — matrix operand order repaired and checked by reading, not compiled, not run on hardware
//
#include <hip/hip_runtime.h>

typedef __attribute__((ext_vector_type(16))) _Float16 v16h;
typedef __attribute__((ext_vector_type(8)))  _Float16 h8;
typedef __attribute__((ext_vector_type(8)))  float    v8f;
typedef __attribute__((ext_vector_type(4)))  float    f32x4;

#define QSZ 6389760ull   // 8*8*1040*96 floats (one of q/k/v)

// ---------------- WMMA fragment helpers (CDNA5 16x16x32 f16 layouts) -------
// A 16x32: lanes 0-15 -> M=lane, K in {half*8..half*8+7, 16+half*8..+7}
__device__ __forceinline__ v16h frag_a(const _Float16* row, int half) {
  h8 lo = *(const h8*)(row + half * 8);
  h8 hi = *(const h8*)(row + 16 + half * 8);
  v16h r;
#pragma unroll
  for (int e = 0; e < 8; ++e) { r[e] = lo[e]; r[8 + e] = hi[e]; }
  return r;
}
// B 32x16 stored transposed in LDS (row n, 32 K halves): lanes 0-15 -> N=lane,
// K in {half*8..half*8+7, 16+half*8..+7} (same as A).
__device__ __forceinline__ v16h frag_b(const _Float16* row, int half) {
  h8 lo = *(const h8*)(row + half * 8);
  h8 hi = *(const h8*)(row + 16 + half * 8);
  v16h r;
#pragma unroll
  for (int e = 0; e < 8; ++e) { r[e] = lo[e]; r[8 + e] = hi[e]; }
  return r;
}
__device__ __forceinline__ void ld_cvt8(_Float16* dst, const float* src) {
  f32x4 a = *(const f32x4*)src;
  f32x4 b = *(const f32x4*)(src + 4);
  h8 h;
  h[0] = (_Float16)a[0]; h[1] = (_Float16)a[1]; h[2] = (_Float16)a[2]; h[3] = (_Float16)a[3];
  h[4] = (_Float16)b[0]; h[5] = (_Float16)b[1]; h[6] = (_Float16)b[2]; h[7] = (_Float16)b[3];
  *(h8*)dst = h;
}
__device__ __forceinline__ void ld_cvt8_guard(_Float16* dst, const float* src,
                                              bool rowok, int kvalid) {
#pragma unroll
  for (int e = 0; e < 8; ++e)
    dst[e] = (rowok && e < kvalid) ? (_Float16)src[e] : (_Float16)0.0f;
}

// ---------------- Generic WMMA GEMM, 128x64 tile, BK=32, 8 waves -----------
// MODE 0: qkv   A=concat(tp,x) [8320x768], Bt=qkv_w[2304][768], scatter q/k/v
// MODE 1: q@kT  per z=(b*8+h): A=q[z] 1040x96, Bt=k[z] 1040x96 -> raw_spa
// MODE 2: att@v per z: A=spa_attn[z] 1040x1040 (lda 1040), B=v[z] [k][96]
// MODE 3: proj  A=attnout 8320x768, Bt=proj_w, +bias, split x_out / tp_out
// MODE 4: plain A[M][lda], Bt=W[N][K], +bias, C[m*ldc+n]
template <int MODE>
__global__ __launch_bounds__(256) void wmma_gemm(
    const float* __restrict__ A0, const float* __restrict__ A1,
    const float* __restrict__ Bw, const float* __restrict__ bias,
    float* __restrict__ C0, float* __restrict__ C1,
    int M, int N, int K, int lda, int ldc) {
  __shared__ __align__(16) _Float16 As[128 * 40];
  __shared__ __align__(16) _Float16 Bs[64 * 40];
  const int tid = threadIdx.x;
  const int lane = tid & 31, wave = tid >> 5;
  const int waveM = wave & 3, waveN = wave >> 2;
  const int m0 = blockIdx.x * 128, n0 = blockIdx.y * 64;
  const int z = blockIdx.z;
  const int lrow = lane & 15, lhalf = lane >> 4;

  v8f acc[2][2] = {};

  for (int k0 = 0; k0 < K; k0 += 32) {
    // ---- A tile: 128x32 halves, 512 8-wide segments over 256 threads ----
#pragma unroll
    for (int it = 0; it < 2; ++it) {
      int s = tid + it * 256;
      int row = s >> 2, cs = (s & 3) << 3;
      int m = m0 + row, kk = k0 + cs;
      _Float16* dst = &As[row * 40 + cs];
      const float* src;
      bool mok = (m < M);
      if (MODE == 0) {
        int b = m / 1040, n = m % 1040;
        src = (n < 16) ? (A1 + ((size_t)(b * 16 + n)) * 768 + kk)
                       : (A0 + ((size_t)b * 1024 + (n - 16)) * 768 + kk);
      } else if (MODE == 1) {
        src = A0 + (size_t)z * (1040 * 96) + (size_t)m * 96 + kk;
      } else if (MODE == 2) {
        src = A0 + (size_t)z * (1040ull * 1040ull) + (size_t)m * 1040 + kk;
      } else {
        src = A0 + (size_t)m * lda + kk;
      }
      int kv = K - kk;
      if (mok && kv >= 8) ld_cvt8(dst, src);
      else                ld_cvt8_guard(dst, src, mok, kv);
    }
    // ---- B tile (transposed rows of 32 K halves) ----
    if (MODE == 2) {  // V stored [k][96]: transpose while staging
      int kr = tid >> 3, nc = (tid & 7) << 3;
      int kk = k0 + kr;
      const float* src = A1 + (size_t)z * (1040 * 96) + (size_t)kk * 96 + (n0 + nc);
#pragma unroll
      for (int e = 0; e < 8; ++e) {
        int n = n0 + nc + e;
        float fv = (kk < K && n < 96) ? src[e] : 0.0f;
        Bs[(nc + e) * 40 + kr] = (_Float16)fv;
      }
    } else {
      int row = tid >> 2, cs = (tid & 3) << 3;
      int n = n0 + row, kk = k0 + cs;
      const float* src;
      if (MODE == 1) src = Bw + (size_t)z * (1040 * 96) + (size_t)n * 96 + kk;
      else           src = Bw + (size_t)n * K + kk;
      _Float16* dst = &Bs[row * 40 + cs];
      if (n < N) ld_cvt8(dst, src);
      else       ld_cvt8_guard(dst, src, false, 8);
    }
    __syncthreads();

    v16h af[2], bf[2];
#pragma unroll
    for (int i = 0; i < 2; ++i)
      af[i] = frag_a(&As[(waveM * 32 + i * 16 + lrow) * 40], lhalf);
#pragma unroll
    for (int j = 0; j < 2; ++j)
      bf[j] = frag_b(&Bs[(waveN * 32 + j * 16 + lrow) * 40], lhalf);
#pragma unroll
    for (int i = 0; i < 2; ++i)
#pragma unroll
      for (int j = 0; j < 2; ++j)
        acc[i][j] = __builtin_amdgcn_wmma_f32_16x16x32_f16(
            false, af[i], false, bf[j], (short)0, acc[i][j], false, false);
    __syncthreads();
  }

  // ---- epilogue: C/D layout -> m = tile_m + 8*(lane>=16) + r, n = lane&15 --
#pragma unroll
  for (int i = 0; i < 2; ++i) {
#pragma unroll
    for (int j = 0; j < 2; ++j) {
      int tm = m0 + waveM * 32 + i * 16;
      int tn = n0 + waveN * 32 + j * 16;
      int n = tn + lrow;
      int mb = tm + lhalf * 8;
#pragma unroll
      for (int r = 0; r < 8; ++r) {
        int m = mb + r;
        if (m >= M || n >= N) continue;
        float val = acc[i][j][r];
        if (MODE == 0) {
          int b = m / 1040, nn = m % 1040;
          int which = n / 768, rem = n % 768;
          int h = rem / 96, d = rem % 96;
          C0[(size_t)which * QSZ + (((size_t)(b * 8 + h)) * 1040 + nn) * 96 + d] = val;
        } else if (MODE == 1) {
          C0[(size_t)z * (1040ull * 1040ull) + (size_t)m * 1040 + n] = val;
        } else if (MODE == 2) {
          int b = z >> 3, h = z & 7;
          C0[((size_t)b * 1040 + m) * 768 + h * 96 + n] = val;
        } else if (MODE == 3) {
          val += bias[n];
          int b = m / 1040, nn = m % 1040;
          if (nn < 16) C1[((size_t)(b * 16 + nn)) * 768 + n] = val;
          else         C0[((size_t)b * 1024 + (nn - 16)) * 768 + n] = val;
        } else {
          C0[(size_t)m * ldc + n] = val + bias[n];
        }
      }
    }
  }
}

// ---------------- windowed channel attention: per (b,w): 16x768, K=64 ------
__global__ __launch_bounds__(256) void chan_gemm(const float* __restrict__ x,
                                                 const float* __restrict__ cp,
                                                 float* __restrict__ out_raw) {
  __shared__ __align__(16) _Float16 As[16 * 72];
  __shared__ __align__(16) _Float16 Bs[128 * 72];
  const int bw = blockIdx.x;                 // 0..127
  const int b = bw >> 4, w = bw & 15;
  const int nh = w >> 2, nw = w & 3;
  const int tid = threadIdx.x, wave = tid >> 5, lane = tid & 31;
  const int lrow = lane & 15, lhalf = lane >> 4;
  {  // A: qw[t][d] = chan_prompts[b,t,pix(w,d)]
    int t = tid >> 4;
    int d0 = (tid & 15) * 4;
    int i = d0 >> 3, j0 = d0 & 7;
    int pix = (nh * 8 + i) * 32 + nw * 8 + j0;
    const float* src = cp + ((size_t)(b * 16 + t)) * 1024 + pix;
#pragma unroll
    for (int e = 0; e < 4; ++e) As[t * 72 + d0 + e] = (_Float16)src[e];
  }
  for (int nc0 = 0; nc0 < 768; nc0 += 128) {
    __syncthreads();
    {  // B: kw[c][d] = x[b, pix(w,d), c], transposed into LDS
      int d = tid >> 2;
      int cs = (tid & 3) * 32;
      int i = d >> 3, j = d & 7;
      int pix = (nh * 8 + i) * 32 + nw * 8 + j;
      const float* src = x + ((size_t)b * 1024 + pix) * 768 + nc0 + cs;
#pragma unroll
      for (int e = 0; e < 32; ++e) Bs[(cs + e) * 72 + d] = (_Float16)src[e];
    }
    __syncthreads();
    v8f acc = {};
    int tn = wave * 16;
#pragma unroll
    for (int kc = 0; kc < 2; ++kc) {
      v16h af = frag_a(&As[lrow * 72 + kc * 32], lhalf);
      v16h bf = frag_b(&Bs[(tn + lrow) * 72 + kc * 32], lhalf);
      acc = __builtin_amdgcn_wmma_f32_16x16x32_f16(false, af, false, bf,
                                                   (short)0, acc, false, false);
    }
    int c = nc0 + tn + lrow;
#pragma unroll
    for (int rr = 0; rr < 8; ++rr) {
      int t = lhalf * 8 + rr;
      out_raw[(((size_t)(b * 16 + t)) * 768 + c) * 16 + w] = acc[rr];
    }
  }
}

// ---------------- softmax over 1040 (spatial attention) --------------------
__global__ __launch_bounds__(256) void softmax_spa(const float* __restrict__ raw,
                                                   float* __restrict__ attn) {
  __shared__ float red[256];
  const size_t row = blockIdx.x;  // 8*8*1040 rows
  const float scale = 0.10206207261596575f;  // 96^-0.5
  const float* r = raw + row * 1040;
  float* o = attn + row * 1040;
  int tid = threadIdx.x;
  float vals[5];
  float mx = -3.4e38f;
#pragma unroll
  for (int i = 0; i < 5; ++i) {
    int c = tid + i * 256;
    if (c < 1040) { float v = r[c] * scale; vals[i] = v; mx = fmaxf(mx, v); }
  }
  red[tid] = mx; __syncthreads();
  for (int s = 128; s > 0; s >>= 1) {
    if (tid < s) red[tid] = fmaxf(red[tid], red[tid + s]);
    __syncthreads();
  }
  mx = red[0]; __syncthreads();
  float sum = 0.f;
#pragma unroll
  for (int i = 0; i < 5; ++i) {
    int c = tid + i * 256;
    if (c < 1040) { float e = __expf(vals[i] - mx); vals[i] = e; sum += e; }
  }
  red[tid] = sum; __syncthreads();
  for (int s = 128; s > 0; s >>= 1) {
    if (tid < s) red[tid] += red[tid + s];
    __syncthreads();
  }
  float inv = 1.0f / red[0];
#pragma unroll
  for (int i = 0; i < 5; ++i) {
    int c = tid + i * 256;
    if (c < 1040) o[c] = vals[i] * inv;
  }
}

// ---------------- softmax over c=768, stride-16 (channel attention) --------
__global__ __launch_bounds__(256) void softmax_chan(const float* __restrict__ raw_r,
                                                    float* __restrict__ attn_r) {
  __shared__ float red[256];
  const int row = blockIdx.x;  // 2048 rows: (b,t,w)
  const int w = row & 15, bt = row >> 4;
  const size_t base = (size_t)bt * 768 * 16 + w;
  int tid = threadIdx.x;
  float vals[3];
  float mx = -3.4e38f;
#pragma unroll
  for (int i = 0; i < 3; ++i) {
    int c = tid + i * 256;
    float v = raw_r[base + (size_t)c * 16] * 0.125f;  // 64^-0.5
    vals[i] = v; mx = fmaxf(mx, v);
  }
  red[tid] = mx; __syncthreads();
  for (int s = 128; s > 0; s >>= 1) {
    if (tid < s) red[tid] = fmaxf(red[tid], red[tid + s]);
    __syncthreads();
  }
  mx = red[0]; __syncthreads();
  float sum = 0.f;
#pragma unroll
  for (int i = 0; i < 3; ++i) { float e = __expf(vals[i] - mx); vals[i] = e; sum += e; }
  red[tid] = sum; __syncthreads();
  for (int s = 128; s > 0; s >>= 1) {
    if (tid < s) red[tid] += red[tid + s];
    __syncthreads();
  }
  float inv = 1.0f / red[0];
#pragma unroll
  for (int i = 0; i < 3; ++i) {
    int c = tid + i * 256;
    attn_r[base + (size_t)c * 16] = vals[i] * inv;
  }
}

// ---------------- quaternion Hamilton-product weight ----------------------
// w[row][col][dk], row in 0..7 (2 per component), col in 0..3
__device__ __forceinline__ float quat_w(const float* qr, const float* qi,
                                        const float* qj, const float* qk,
                                        int row, int col, int dk) {
  const int q = row >> 1, rp = row & 1;
  const int sel_t[4][4]   = {{0, 1, 2, 3}, {1, 0, 3, 2}, {2, 3, 0, 1}, {3, 2, 1, 0}};
  const float sgn_t[4][4] = {{1, -1, -1, -1}, {1, 1, -1, 1}, {1, 1, 1, -1}, {1, -1, 1, 1}};
  const float* src;
  switch (sel_t[q][col]) {
    case 0:  src = qr; break;
    case 1:  src = qi; break;
    case 2:  src = qj; break;
    default: src = qk; break;
  }
  return sgn_t[q][col] * src[rp * 3 + dk];
}

__global__ __launch_bounds__(256) void quat_conv1(
    const float* __restrict__ tp, const float* __restrict__ tp_out,
    const float* __restrict__ ch_p, const float* __restrict__ qr,
    const float* __restrict__ qi, const float* __restrict__ qj,
    const float* __restrict__ qk, const float* __restrict__ qb,
    float* __restrict__ fe1) {
  int idx = blockIdx.x * 256 + threadIdx.x;  // < 98304
  int bt = idx / 768, c = idx % 768;
  float acc[8];
#pragma unroll
  for (int oc = 0; oc < 8; ++oc) acc[oc] = qb[oc];
  const float* srcs[3] = {tp + (size_t)bt * 768, tp_out + (size_t)bt * 768,
                          ch_p + (size_t)bt * 768};
#pragma unroll
  for (int s = 0; s < 3; ++s) {
    int ic = s + 1;  // ic==0 input is all-zero
    const float* xp = srcs[s];
    float xm = (c > 0) ? xp[c - 1] : 0.f;
    float x0 = xp[c];
    float xl = (c < 767) ? xp[c + 1] : 0.f;
#pragma unroll
    for (int oc = 0; oc < 8; ++oc)
      acc[oc] += xm * quat_w(qr, qi, qj, qk, oc, ic, 0) +
                 x0 * quat_w(qr, qi, qj, qk, oc, ic, 1) +
                 xl * quat_w(qr, qi, qj, qk, oc, ic, 2);
  }
#pragma unroll
  for (int oc = 0; oc < 8; ++oc) fe1[((size_t)bt * 8 + oc) * 768 + c] = acc[oc];
}

__global__ __launch_bounds__(256) void quat_conv2(
    const float* __restrict__ fe1, const float* __restrict__ dr,
    const float* __restrict__ di, const float* __restrict__ dj,
    const float* __restrict__ dkk, const float* __restrict__ db,
    float* __restrict__ fe2) {
  int idx = blockIdx.x * 256 + threadIdx.x;
  int bt = idx / 768, c = idx % 768;
  float acc[4];
#pragma unroll
  for (int oc = 0; oc < 4; ++oc) acc[oc] = db[oc];
#pragma unroll
  for (int ic = 0; ic < 8; ++ic) {
    const float* xp = fe1 + ((size_t)bt * 8 + ic) * 768;
    float xm = (c > 0) ? xp[c - 1] : 0.f;
    float x0 = xp[c];
    float xl = (c < 767) ? xp[c + 1] : 0.f;
    // w2c[oc][ic][dk] = w2[ic][oc][2-dk]  (flip + transpose)
#pragma unroll
    for (int oc = 0; oc < 4; ++oc)
      acc[oc] += xm * quat_w(dr, di, dj, dkk, ic, oc, 2) +
                 x0 * quat_w(dr, di, dj, dkk, ic, oc, 1) +
                 xl * quat_w(dr, di, dj, dkk, ic, oc, 0);
  }
#pragma unroll
  for (int oc = 0; oc < 4; ++oc) fe2[((size_t)bt * 4 + oc) * 768 + c] = acc[oc];
}

__global__ __launch_bounds__(256) void bn_stats(const float* __restrict__ fe2,
                                                float* __restrict__ st) {
  __shared__ float s1[256], s2[256];
  const int g = blockIdx.x;  // b*4+oc, 32 groups
  const int b = g >> 2, oc = g & 3;
  int tid = threadIdx.x;
  float a = 0.f, q = 0.f;
  for (int i = 0; i < 48; ++i) {
    int e = tid + i * 256;  // < 12288
    int t = e / 768, c = e % 768;
    float v = fe2[(((size_t)(b * 16 + t)) * 4 + oc) * 768 + c];
    a += v; q += v * v;
  }
  s1[tid] = a; s2[tid] = q; __syncthreads();
  for (int s = 128; s > 0; s >>= 1) {
    if (tid < s) { s1[tid] += s1[tid + s]; s2[tid] += s2[tid + s]; }
    __syncthreads();
  }
  if (tid == 0) {
    float m = s1[0] / 12288.f;
    st[g * 2] = m;
    st[g * 2 + 1] = s2[0] / 12288.f - m * m;
  }
}

__global__ __launch_bounds__(256) void quat_final(
    const float* __restrict__ fe2, const float* __restrict__ st,
    const float* __restrict__ gg, const float* __restrict__ bb,
    const float* __restrict__ qfto, float* __restrict__ outq) {
  int idx = blockIdx.x * 256 + threadIdx.x;
  int bt = idx / 768, c = idx % 768;
  int b = bt >> 4;
  float acc = 0.f;
#pragma unroll
  for (int oc = 0; oc < 4; ++oc) {
    float m = st[(b * 4 + oc) * 2];
    float v = st[(b * 4 + oc) * 2 + 1];
    float xv = fe2[((size_t)bt * 4 + oc) * 768 + c];
    float xn = (xv - m) * rsqrtf(v + 1e-5f) * gg[oc] + bb[oc];
    float ge = 0.5f * xn * (1.0f + erff(xn * 0.70710678118654752f));
    acc += ge * qfto[oc];
  }
  outq[idx] = acc;
}

// ---------------------------------------------------------------------------
extern "C" void kernel_launch(void* const* d_in, const int* in_sizes, int n_in,
                              void* d_out, int out_size, void* d_ws, size_t ws_size,
                              hipStream_t stream) {
  const float* x      = (const float*)d_in[0];   // (8,1024,768)
  const float* tp     = (const float*)d_in[1];   // (8,16,768)
  const float* qkv_w  = (const float*)d_in[2];   // (2304,768)
  const float* proj_w = (const float*)d_in[3];   // (768,768)
  const float* proj_b = (const float*)d_in[4];   // (768)
  const float* tt_w   = (const float*)d_in[5];   // (1024,768)
  const float* tt_b   = (const float*)d_in[6];   // (1024)
  const float* tt1_w  = (const float*)d_in[7];   // (768,1024)
  const float* tt1_b  = (const float*)d_in[8];   // (768)
  const float* qf_r   = (const float*)d_in[9];
  const float* qf_i   = (const float*)d_in[10];
  const float* qf_j   = (const float*)d_in[11];
  const float* qf_k   = (const float*)d_in[12];
  const float* qf_b   = (const float*)d_in[13];
  const float* dqf_r  = (const float*)d_in[14];
  const float* dqf_i  = (const float*)d_in[15];
  const float* dqf_j  = (const float*)d_in[16];
  const float* dqf_k  = (const float*)d_in[17];
  const float* dqf_b  = (const float*)d_in[18];
  const float* bn_g   = (const float*)d_in[19];
  const float* bn_b   = (const float*)d_in[20];
  const float* qfto_w = (const float*)d_in[21];  // (1,4,1)

  float* out = (float*)d_out;
  float* ws  = (float*)d_ws;

  // workspace layout (floats)
  float* qbuf    = ws;                    // [b,h,n,96]
  float* kbuf    = ws + QSZ;
  float* vbuf    = ws + 2 * QSZ;
  float* attnout = ws + 3 * QSZ;          // [b,n,768]
  float* tp_out  = ws + 4 * QSZ;          // 98304
  float* chan_p  = tp_out + 98304;        // 131072
  float* ch_p    = chan_p + 131072;       // 98304
  float* fe1     = ch_p + 98304;          // 786432
  float* fe2     = fe1 + 786432;          // 393216
  float* bnst    = fe2 + 393216;          // 64

  // output layout (floats, return order)
  float* o_xout   = out;                  // 8*1024*768
  float* o_raw    = out + 6291456;        // 8*8*1040*1040
  float* o_attn   = out + 75513856;
  float* o_rawch  = out + 144736256;      // 8*16*768*16
  float* o_attnch = out + 146309120;
  float* o_quat   = out + 147881984;      // 8*16*768

  dim3 blk(256);

  // 1) QKV projection with fused concat + [b,h,n,d] scatter
  wmma_gemm<0><<<dim3(65, 36, 1), blk, 0, stream>>>(
      x, tp, qkv_w, nullptr, qbuf, nullptr, 8320, 2304, 768, 768, 0);
  // 2) raw_spa = q @ k^T per (b,h)
  wmma_gemm<1><<<dim3(9, 17, 64), blk, 0, stream>>>(
      qbuf, nullptr, kbuf, nullptr, o_raw, nullptr, 1040, 1040, 96, 96, 0);
  // 3) softmax (scale 96^-0.5)
  softmax_spa<<<66560, blk, 0, stream>>>(o_raw, o_attn);
  // 4) attn @ v
  wmma_gemm<2><<<dim3(9, 2, 64), blk, 0, stream>>>(
      o_attn, vbuf, nullptr, nullptr, attnout, nullptr, 1040, 96, 1040, 1040, 0);
  // 5) output projection, split tp_out / x_out
  wmma_gemm<3><<<dim3(65, 12, 1), blk, 0, stream>>>(
      attnout, nullptr, proj_w, proj_b, o_xout, tp_out, 8320, 768, 768, 768, 0);
  // 6) chan_prompts = tp @ tt_w^T + tt_b
  wmma_gemm<4><<<dim3(1, 16, 1), blk, 0, stream>>>(
      tp, nullptr, tt_w, tt_b, chan_p, nullptr, 128, 1024, 768, 768, 1024);
  // 7) windowed channel attention logits (scattered to raw_chan_r layout)
  chan_gemm<<<128, blk, 0, stream>>>(x, chan_p, o_rawch);
  // 8) channel softmax over c=768 (stride-16 in r layout)
  softmax_chan<<<2048, blk, 0, stream>>>(o_rawch, o_attnch);
  // 9) ch_prompts = chan_prompts @ tt1_w^T + tt1_b
  wmma_gemm<4><<<dim3(1, 12, 1), blk, 0, stream>>>(
      chan_p, nullptr, tt1_w, tt1_b, ch_p, nullptr, 128, 768, 1024, 1024, 768);
  // 10) quaternion feature path
  quat_conv1<<<384, blk, 0, stream>>>(tp, tp_out, ch_p, qf_r, qf_i, qf_j, qf_k,
                                      qf_b, fe1);
  quat_conv2<<<384, blk, 0, stream>>>(fe1, dqf_r, dqf_i, dqf_j, dqf_k, dqf_b, fe2);
  bn_stats<<<32, blk, 0, stream>>>(fe2, bnst);
  quat_final<<<384, blk, 0, stream>>>(fe2, bnst, bn_g, bn_b, qfto_w, o_quat);
}
